// GAT_9990093931281
// MI455X (gfx1250) — matrix operand order repaired and checked by reading, not compiled, not run on hardware
//
#include <hip/hip_runtime.h>
#include <hip/hip_bf16.h>
#include <math.h>

typedef __attribute__((ext_vector_type(16))) _Float16 v16h;
typedef __attribute__((ext_vector_type(8)))  float    v8f;

#define GAT_NEG_SLOPE 0.2f

// ---------------------------------------------------------------- utilities
__global__ void k_fill_f32(float* __restrict__ p, float v, long n) {
  long i = (long)blockIdx.x * blockDim.x + threadIdx.x;
  if (i < n) p[i] = v;
}
__global__ void k_fill_u32(unsigned* __restrict__ p, unsigned v, long n) {
  long i = (long)blockIdx.x * blockDim.x + threadIdx.x;
  if (i < n) p[i] = v;
}

// ---------------------------------------------------------------- packing
// Pack f32 source directly into WMMA f16 fragment order so the GEMM does
// one 32B contiguous load per operand per K-step.
//
// A fragment (16x32 f16, ISA 7.12.2): lane%16 = row, lane/16 selects K-halves:
//   lanes 0..15 : V0..7 -> K 0..15 (contiguous, same order as B)
//   lanes 16..31: V0..7 -> K 16..31
// Apk layout: [tileM][kt][lane][16 halves], contiguous per lane.
__global__ void k_pack_a(const float* __restrict__ X, _Float16* __restrict__ Apk,
                         int M, int K) {
  int KT = K >> 5;
  long t = (long)blockIdx.x * blockDim.x + threadIdx.x;
  long tot = (long)(M >> 4) * KT * 32;
  if (t >= tot) return;
  int  lane = (int)(t & 31);
  long r    = t >> 5;
  int  kt   = (int)(r % KT);
  int  tm   = (int)(r / KT);
  int  row  = tm * 16 + (lane & 15);
  int  akoff = (lane >> 4) * 16;
  const float* src = X + (long)row * K + kt * 32;
  v16h frag;
#pragma unroll
  for (int v = 0; v < 8; ++v) {
    int ka = akoff + 2 * v;
    frag[2 * v]     = (_Float16)src[ka];
    frag[2 * v + 1] = (_Float16)src[ka + 1];
  }
  *(v16h*)(Apk + t * 16) = frag;
}

// B fragment (32x16 f16): lane%16 = column, lanes 0..15 hold K 0..15
// (2 per VGPR), lanes 16..31 hold K 16..31.
// Bpk layout: [tileN][kt][lane][16 halves].
__global__ void k_pack_b(const float* __restrict__ W, _Float16* __restrict__ Bpk,
                         int K, int Nn) {
  int KT = K >> 5;
  long t = (long)blockIdx.x * blockDim.x + threadIdx.x;
  long tot = (long)(Nn >> 4) * KT * 32;
  if (t >= tot) return;
  int  lane = (int)(t & 31);
  long r    = t >> 5;
  int  kt   = (int)(r % KT);
  int  tn   = (int)(r / KT);
  int  col  = tn * 16 + (lane & 15);
  int  kb0  = kt * 32 + (lane >> 4) * 16;
  v16h frag;
#pragma unroll
  for (int v = 0; v < 8; ++v) {
    frag[2 * v]     = (_Float16)W[(long)(kb0 + 2 * v) * Nn + col];
    frag[2 * v + 1] = (_Float16)W[(long)(kb0 + 2 * v + 1) * Nn + col];
  }
  *(v16h*)(Bpk + t * 16) = frag;
}

// ---------------------------------------------------------------- WMMA GEMM
// C[M,Nn] = A*B with pre-packed fragments: per K-step each lane does two
// contiguous 32B loads + one v_wmma_f32_16x16x32_f16. One wave per 16x16 tile.
__global__ void k_wmma_gemm_packed(const _Float16* __restrict__ Apk,
                                   const _Float16* __restrict__ Bpk,
                                   float* __restrict__ C,
                                   int M, int Nn, int K) {
  int KT     = K >> 5;
  int tilesN = Nn >> 4;
  int tiles  = (M >> 4) * tilesN;
  int wave   = blockIdx.x * (blockDim.x >> 5) + (threadIdx.x >> 5);
  if (wave >= tiles) return;                 // uniform per wave; EXEC stays all-1s
  int tm = wave / tilesN;
  int tn = wave - tm * tilesN;
  int lane = threadIdx.x & 31;

  const v16h* Ap = (const v16h*)Apk + (long)tm * KT * 32 + lane;
  const v16h* Bp = (const v16h*)Bpk + (long)tn * KT * 32 + lane;

  v8f acc = {};
#pragma unroll 4
  for (int kt = 0; kt < KT; ++kt) {
    v16h a = Ap[(long)kt * 32];
    v16h b = Bp[(long)kt * 32];
    acc = __builtin_amdgcn_wmma_f32_16x16x32_f16(
        false, a, false, b, (short)0, acc, false, false);
  }
  // C/D: lane%16 = column, VGPR v -> row = v + 8*(lane/16)
  float* Cp = C + (long)(tm * 16) * Nn + tn * 16 + (lane & 15);
  int rbase = (lane >> 4) * 8;
#pragma unroll
  for (int v = 0; v < 8; ++v) Cp[(long)(rbase + v) * Nn] = acc[v];
}

// ---------------------------------------------------------------- attention
__device__ __forceinline__ unsigned ordkey(float f) {
  unsigned u = __float_as_uint(f);
  return (u & 0x80000000u) ? ~u : (u | 0x80000000u);
}
__device__ __forceinline__ float ordval(unsigned u) {
  return __uint_as_float((u & 0x80000000u) ? (u & 0x7fffffffu) : ~u);
}
__device__ __forceinline__ void get_edge(const int* __restrict__ ei, int E, long e,
                                         int& s, int& d) {
  if (e < E) { s = ei[e]; d = ei[E + e]; } else { s = d = (int)(e - E); }
}

__global__ void k_node_scores(const float* __restrict__ h,
                              const float* __restrict__ a_src,
                              const float* __restrict__ a_dst,
                              float* __restrict__ es, float* __restrict__ ed,
                              int n, int heads, int ch) {
  long t = (long)blockIdx.x * blockDim.x + threadIdx.x;
  if (t >= (long)n * heads) return;
  int node = (int)(t / heads), hh = (int)(t % heads);
  const float* hp = h + ((long)node * heads + hh) * ch;
  const float* as = a_src + (long)hh * ch;
  const float* ad = a_dst + (long)hh * ch;
  float s0 = 0.f, s1 = 0.f;
  for (int c = 0; c < ch; c += 4) {
    float4 v = *(const float4*)(hp + c);
    float4 a = *(const float4*)(as + c);
    float4 b = *(const float4*)(ad + c);
    s0 += v.x * a.x + v.y * a.y + v.z * a.z + v.w * a.w;
    s1 += v.x * b.x + v.y * b.y + v.z * b.z + v.w * b.w;
  }
  es[t] = s0; ed[t] = s1;
}

__global__ void k_edge_max(const int* __restrict__ ei, int E, int n, int heads,
                           const float* __restrict__ es, const float* __restrict__ ed,
                           unsigned* __restrict__ m) {
  long t = (long)blockIdx.x * blockDim.x + threadIdx.x;
  long tot = (long)(E + n) * heads;
  if (t >= tot) return;
  long e = t / heads; int hh = (int)(t - e * heads);
  int s, d; get_edge(ei, E, e, s, d);
  float x = es[(long)s * heads + hh] + ed[(long)d * heads + hh];
  x = (x > 0.f) ? x : GAT_NEG_SLOPE * x;
  atomicMax(&m[(long)d * heads + hh], ordkey(x));
}

__global__ void k_edge_exp(const int* __restrict__ ei, int E, int n, int heads,
                           const float* __restrict__ es, const float* __restrict__ ed,
                           const unsigned* __restrict__ m,
                           float* __restrict__ p, float* __restrict__ ssum) {
  long t = (long)blockIdx.x * blockDim.x + threadIdx.x;
  long tot = (long)(E + n) * heads;
  if (t >= tot) return;
  long e = t / heads; int hh = (int)(t - e * heads);
  int s, d; get_edge(ei, E, e, s, d);
  float x = es[(long)s * heads + hh] + ed[(long)d * heads + hh];
  x = (x > 0.f) ? x : GAT_NEG_SLOPE * x;
  float mv = ordval(m[(long)d * heads + hh]);
  if (mv == -INFINITY) mv = 0.f;     // empty-segment rule from reference
  float pe = __expf(x - mv);
  p[t] = pe;
  atomicAdd(&ssum[(long)d * heads + hh], pe);
}

__global__ void k_edge_aggregate(const int* __restrict__ ei, int E, int n, int heads,
                                 int ch, const float* __restrict__ p,
                                 const float* __restrict__ ssum,
                                 const float* __restrict__ feat,
                                 float* __restrict__ outp) {
  long t = (long)blockIdx.x * blockDim.x + threadIdx.x;
  long tot = (long)(E + n) * heads;
  if (t >= tot) return;
  long e = t / heads; int hh = (int)(t - e * heads);
  int s, d; get_edge(ei, E, e, s, d);
  float alpha = p[t] / (ssum[(long)d * heads + hh] + 1e-16f);
  const float* f = feat + ((long)s * heads + hh) * ch;
  float* o = outp + ((long)d * heads + hh) * ch;
  for (int c = 0; c < ch; c += 4) {
    float4 v = *(const float4*)(f + c);
    atomicAdd(o + c + 0, alpha * v.x);
    atomicAdd(o + c + 1, alpha * v.y);
    atomicAdd(o + c + 2, alpha * v.z);
    atomicAdd(o + c + 3, alpha * v.w);
  }
}

__global__ void k_bias_relu(float* __restrict__ h, const float* __restrict__ bias,
                            int n, int dch) {
  long t = (long)blockIdx.x * blockDim.x + threadIdx.x;
  if (t >= (long)n * dch) return;
  int j = (int)(t % dch);
  float v = h[t] + bias[j];
  h[t] = v > 0.f ? v : 0.f;
}

__global__ void k_add_bias(float* __restrict__ o, const float* __restrict__ b,
                           int n, int dch) {
  long t = (long)blockIdx.x * blockDim.x + threadIdx.x;
  if (t >= (long)n * dch) return;
  o[t] += b[t % dch];
}

// ---------------------------------------------------------------- launch
extern "C" void kernel_launch(void* const* d_in, const int* in_sizes, int n_in,
                              void* d_out, int out_size, void* d_ws, size_t ws_size,
                              hipStream_t stream) {
  const int IN = 512, HID = 64, H = 8, D1 = 512, OUT = 128;
  const float* x   = (const float*)d_in[0];
  const int*   ei  = (const int*)d_in[1];
  const float* W1  = (const float*)d_in[2];
  const float* b1  = (const float*)d_in[3];
  const float* as1 = (const float*)d_in[4];
  const float* ad1 = (const float*)d_in[5];
  const float* W2  = (const float*)d_in[6];
  const float* b2  = (const float*)d_in[7];
  const float* as2 = (const float*)d_in[8];
  const float* ad2 = (const float*)d_in[9];
  float* out = (float*)d_out;

  int n  = in_sizes[0] / IN;   // 50000
  int E  = in_sizes[1] / 2;    // 800000
  int ET = E + n;

  // workspace carve-out (256B aligned)
  char* wsp = (char*)d_ws;
  auto carve = [&](size_t bytes) {
    char* p = wsp;
    wsp += (bytes + 255) & ~(size_t)255;
    return p;
  };
  _Float16* a1pk = (_Float16*)carve((size_t)n * IN * 2);
  _Float16* b1pk = (_Float16*)carve((size_t)IN * D1 * 2);
  _Float16* b2pk = (_Float16*)carve((size_t)D1 * OUT * 2);
  _Float16* a2pk = (_Float16*)carve((size_t)n * D1 * 2);
  float*    h1   = (float*)carve((size_t)n * D1 * 4);
  float*    es1  = (float*)carve((size_t)n * H * 4);
  float*    ed1  = (float*)carve((size_t)n * H * 4);
  unsigned* m1   = (unsigned*)carve((size_t)n * H * 4);
  float*    s1   = (float*)carve((size_t)n * H * 4);
  float*    p1   = (float*)carve((size_t)ET * H * 4);
  float*    ag1  = (float*)carve((size_t)n * D1 * 4);
  float*    g2   = (float*)carve((size_t)n * OUT * 4);
  float*    es2  = (float*)carve((size_t)n * 4);
  float*    ed2  = (float*)carve((size_t)n * 4);
  unsigned* m2   = (unsigned*)carve((size_t)n * 4);
  float*    s2   = (float*)carve((size_t)n * 4);
  float*    p2   = (float*)carve((size_t)ET * 4);

  auto nb = [](long tot, int blk) { return (unsigned)((tot + blk - 1) / blk); };
  const unsigned NEG_INF_KEY = 0x007FFFFFu;  // ordkey(-inf)

  // fragment packing (fused f32->f16 conversion)
  k_pack_a<<<nb((long)n * 32, 256), 256, 0, stream>>>(x, a1pk, n, IN);
  k_pack_b<<<nb((long)(D1 / 16) * (IN / 32) * 32, 256), 256, 0, stream>>>(W1, b1pk, IN, D1);
  k_pack_b<<<nb((long)(OUT / 16) * (D1 / 32) * 32, 256), 256, 0, stream>>>(W2, b2pk, D1, OUT);

  // ---- layer 1: h1 = x @ W1  (M=n, N=512, K=512), WMMA
  {
    long tiles = ((long)n / 16) * (D1 / 16);
    k_wmma_gemm_packed<<<nb(tiles * 32, 128), 128, 0, stream>>>(a1pk, b1pk, h1, n, D1, IN);
  }
  k_node_scores<<<nb((long)n * H, 256), 256, 0, stream>>>(h1, as1, ad1, es1, ed1, n, H, HID);
  k_fill_u32<<<nb((long)n * H, 256), 256, 0, stream>>>(m1, NEG_INF_KEY, (long)n * H);
  k_fill_f32<<<nb((long)n * H, 256), 256, 0, stream>>>(s1, 0.f, (long)n * H);
  k_fill_f32<<<nb((long)n * D1, 256), 256, 0, stream>>>(ag1, 0.f, (long)n * D1);
  k_edge_max<<<nb((long)ET * H, 256), 256, 0, stream>>>(ei, E, n, H, es1, ed1, m1);
  k_edge_exp<<<nb((long)ET * H, 256), 256, 0, stream>>>(ei, E, n, H, es1, ed1, m1, p1, s1);
  k_edge_aggregate<<<nb((long)ET * H, 256), 256, 0, stream>>>(ei, E, n, H, HID, p1, s1, h1, ag1);
  k_bias_relu<<<nb((long)n * D1, 256), 256, 0, stream>>>(ag1, b1, n, D1);

  // ---- layer 2: g2 = relu(h) @ W2  (M=n, N=128, K=512), WMMA
  k_pack_a<<<nb((long)n * 32, 256), 256, 0, stream>>>(ag1, a2pk, n, D1);
  {
    long tiles = ((long)n / 16) * (OUT / 16);
    k_wmma_gemm_packed<<<nb(tiles * 32, 128), 128, 0, stream>>>(a2pk, b2pk, g2, n, OUT, D1);
  }
  k_node_scores<<<nb((long)n, 256), 256, 0, stream>>>(g2, as2, ad2, es2, ed2, n, 1, OUT);
  k_fill_u32<<<nb((long)n, 256), 256, 0, stream>>>(m2, NEG_INF_KEY, (long)n);
  k_fill_f32<<<nb((long)n, 256), 256, 0, stream>>>(s2, 0.f, (long)n);
  k_fill_f32<<<nb((long)n * OUT, 256), 256, 0, stream>>>(out, 0.f, (long)n * OUT);
  k_edge_max<<<nb((long)ET, 256), 256, 0, stream>>>(ei, E, n, 1, es2, ed2, m2);
  k_edge_exp<<<nb((long)ET, 256), 256, 0, stream>>>(ei, E, n, 1, es2, ed2, m2, p2, s2);
  k_edge_aggregate<<<nb((long)ET, 256), 256, 0, stream>>>(ei, E, n, 1, OUT, p2, s2, g2, out);
  k_add_bias<<<nb((long)n * OUT, 256), 256, 0, stream>>>(out, b2, n, OUT);
}
